// RNTN_20057497272388
// MI455X (gfx1250) — matrix operand order repaired and checked by reading, not compiled, not run on hardware
//
#include <hip/hip_runtime.h>
#include <hip/hip_bf16.h>
#include <math.h>

#define HID 128
#define HID2 256
#define OUT_DIM 5
#define N_TREES 16
#define N_LEAVES 64
#define NODES_PER_TREE 127   // 2*64 - 1
#define N_NODES 2032         // 16 * 127
#define MAX_H 6              // perfect tree: heights 0..6
#define TILES_PER_BLOCK 4    // 4 waves/block, one node tile each, shared V panel
#define ROW_PITCH 264        // 256 + 8 bf16 pad -> lane banks spread (528B row)

typedef __attribute__((ext_vector_type(16))) __bf16 v16bf;
typedef __attribute__((ext_vector_type(8)))  __bf16 v8bf;
typedef __attribute__((ext_vector_type(8)))  float  v8f;

// ---------------------------------------------------------------------------
// Compile-time tree topology (mirrors reference _build_trees for one tree;
// all 16 trees are identical, global node = tree*127 + local).
// ---------------------------------------------------------------------------
struct TreeT {
    int left[NODES_PER_TREE];
    int right[NODES_PER_TREE];
    int leaf[NODES_PER_TREE];
    int height[NODES_PER_TREE];
};
struct LevelsT {
    int idx[MAX_H + 1][N_LEAVES];
    int cnt[MAX_H + 1];
};

constexpr int tree_rec(TreeT& t, int& c, int nl) {
    if (nl == 1) {
        int i = c++;
        t.leaf[i] = 1; t.left[i] = i; t.right[i] = i; t.height[i] = 0;
        return i;
    }
    int l = tree_rec(t, c, nl / 2);
    int r = tree_rec(t, c, nl - nl / 2);
    int i = c++;
    t.leaf[i] = 0; t.left[i] = l; t.right[i] = r; t.height[i] = t.height[l] + 1;
    return i;
}
constexpr TreeT build_tree() { TreeT t{}; int c = 0; tree_rec(t, c, N_LEAVES); return t; }
constexpr LevelsT build_levels() {
    LevelsT L{};
    TreeT t = build_tree();
    for (int i = 0; i < NODES_PER_TREE; ++i) {
        int h = t.height[i];
        L.idx[h][L.cnt[h]++] = i;
    }
    return L;
}

__constant__ TreeT   g_tree = build_tree();
__constant__ LevelsT g_lv   = build_levels();

// ---------------------------------------------------------------------------
// Kernels
// ---------------------------------------------------------------------------
__global__ void k_cvt_bf16(const float* __restrict__ src, __bf16* __restrict__ dst, int n) {
    int i = blockIdx.x * blockDim.x + threadIdx.x;
    if (i < n) dst[i] = (__bf16)src[i];
}

// Wt[h][l] = W[l][h]  (transpose for contiguous per-h dot in finalize)
__global__ void k_wt(const float* __restrict__ W, float* __restrict__ Wt) {
    int i = blockIdx.x * blockDim.x + threadIdx.x;
    if (i < HID2 * HID) {
        int l = i / HID, h = i % HID;
        Wt[h * HID2 + l] = W[i];
    }
}

// Leaves: H[n] = embed[word_idx[n]]
__global__ void k_leaf(const float* __restrict__ embed, const int* __restrict__ word_idx,
                       float* __restrict__ H) {
    int n = blockIdx.x;        // node 0..2031
    int h = threadIdx.x;       // 0..127
    int local = n % NODES_PER_TREE;
    if (g_tree.leaf[local]) {
        int w = word_idx[n];
        H[(size_t)n * HID + h] = embed[(size_t)w * HID + h];
    }
}

// Gather children into C (fp32 + bf16 copies), level-slot-major [B][256]
__global__ void k_gather(const float* __restrict__ H, float* __restrict__ Cf,
                         __bf16* __restrict__ Cbf, int level) {
    int j = blockIdx.x;        // level slot 0..B-1
    int l = threadIdx.x;       // 0..255
    int cnt = g_lv.cnt[level];
    int tr = j / cnt, pos = j % cnt;
    int local = g_lv.idx[level][pos];
    int base = tr * NODES_PER_TREE;
    int child = (l < HID) ? (base + g_tree.left[local]) : (base + g_tree.right[local]);
    int lh = (l < HID) ? l : (l - HID);
    float c = H[(size_t)child * HID + lh];
    Cf [(size_t)j * HID2 + l] = c;
    Cbf[(size_t)j * HID2 + l] = (__bf16)c;
}

// Bilinear term via WMMA. Block = 4 waves, one h, 4 node tiles.
//   Out[k,b] = sum_l V[h,k,l] * C[b,l]   (A = V[h] panel from LDS, B = C^T tile in regs)
//   xvx[b,h] = sum_k C[b,k] * Out[k,b]   (fp32 dot on the C/D fragment layout)
// V[h] 16-row panels (8KB, contiguous) are cooperatively streamed global->LDS
// with coalesced b128 loads, double-buffered, shared by all 4 waves (4x less
// L2 traffic than per-wave fetch).
__launch_bounds__(128)
__global__ void k_bilinear(const __bf16* __restrict__ Vbf, const __bf16* __restrict__ Cbf,
                           const float* __restrict__ Cf, float* __restrict__ xvx,
                           int numTiles) {
    __shared__ v8bf lds[2][16 * (ROW_PITCH / 8)];   // 2 x 16 rows x 264 bf16 = 16.9KB

    const int wave = threadIdx.x >> 5;
    const int lane = threadIdx.x & 31;
    const int h    = blockIdx.y;            // 0..127
    const int lhi  = (lane >> 4) & 1;       // which 16-lane half
    const int lm   = lane & 15;

    int bt = blockIdx.x * TILES_PER_BLOCK + wave;
    const bool owner = (bt < numTiles);
    if (!owner) bt = numTiles - 1;          // clamp: duplicate compute, gated store
    const int node = bt * 16 + lm;          // this lane's N-column / node slot

    // B fragment preload: lane j element e = C[bt*16 + (j&15)][lt*32 + 16*lhi + e]
    // -> 16 contiguous bf16 per chunk: two b128 loads, kept register-resident.
    const __bf16* crow = Cbf + (size_t)node * HID2 + lhi * 16;
    v16bf breg[8];
#pragma unroll
    for (int lt = 0; lt < 8; ++lt) {
        v8bf b0 = *reinterpret_cast<const v8bf*>(crow + lt * 32);
        v8bf b1 = *reinterpret_cast<const v8bf*>(crow + lt * 32 + 8);
        breg[lt] = __builtin_shufflevector(b0, b1, 0,1,2,3,4,5,6,7,8,9,10,11,12,13,14,15);
    }

    // Prologue: panel for kt=0 into LDS buffer 0 (coalesced: 512 x 16B chunks).
    {
        const v8bf* g = reinterpret_cast<const v8bf*>(Vbf + (size_t)h * HID2 * HID2);
#pragma unroll
        for (int p = 0; p < 4; ++p) {
            int c = threadIdx.x + p * 128;          // chunk 0..511
            lds[0][(c >> 5) * (ROW_PITCH / 8) + (c & 31)] = g[c];
        }
    }
    __syncthreads();

    const float* cfrow = Cf + (size_t)node * HID2;
    float part = 0.f;
    v8bf stage[4];

    for (int kt = 0; kt < 16; ++kt) {
        // Prefetch next panel global->regs (overlaps with WMMA below).
        if (kt < 15) {
            const v8bf* g = reinterpret_cast<const v8bf*>(
                Vbf + ((size_t)h * HID2 + (kt + 1) * 16) * HID2);
#pragma unroll
            for (int p = 0; p < 4; ++p) stage[p] = g[threadIdx.x + p * 128];
        }

        // A fragment from LDS: lane holds row kt*16+lm, K-slots {0..15} (+16 upper half)
        const __bf16* abase =
            reinterpret_cast<const __bf16*>(lds[kt & 1]) + lm * ROW_PITCH + lhi * 16;
        v8f acc = {};
#pragma unroll
        for (int lt = 0; lt < 8; ++lt) {
            v8bf a0 = *reinterpret_cast<const v8bf*>(abase + lt * 32);
            v8bf a1 = *reinterpret_cast<const v8bf*>(abase + lt * 32 + 8);
            v16bf a = __builtin_shufflevector(a0, a1, 0,1,2,3,4,5,6,7,8,9,10,11,12,13,14,15);
            acc = __builtin_amdgcn_wmma_f32_16x16x32_bf16(
                /*neg_a=*/false, a, /*neg_b=*/false, breg[lt],
                /*c_mod=*/(short)0, acc, /*reuse_a=*/false, /*reuse_b=*/false);
        }

        // D layout: lane holds N=lm, M rows kt*16 + r + 8*lhi -> fp32 dot with C[b,k]
        const int koff = kt * 16 + lhi * 8;
        const float4 c0 = *reinterpret_cast<const float4*>(cfrow + koff);
        const float4 c1 = *reinterpret_cast<const float4*>(cfrow + koff + 4);
        part += acc[0]*c0.x + acc[1]*c0.y + acc[2]*c0.z + acc[3]*c0.w
              + acc[4]*c1.x + acc[5]*c1.y + acc[6]*c1.z + acc[7]*c1.w;

        // Drain prefetch regs into the other LDS buffer.
        if (kt < 15) {
#pragma unroll
            for (int p = 0; p < 4; ++p) {
                int c = threadIdx.x + p * 128;
                lds[(kt + 1) & 1][(c >> 5) * (ROW_PITCH / 8) + (c & 31)] = stage[p];
            }
        }
        __syncthreads();
    }

    // combine the two M-halves (lanes j <-> j+16 share the same node column)
    part += __shfl_xor(part, 16, 32);
    if (owner && lane < 16) xvx[(size_t)node * HID + h] = part;
}

// H[node] = tanh(xVx + c@W + b)   (fp32)
__global__ void k_finalize(const float* __restrict__ xvx, const float* __restrict__ Cf,
                           const float* __restrict__ Wt, const float* __restrict__ bias,
                           float* __restrict__ H, int level) {
    int j = blockIdx.x;        // level slot
    int h = threadIdx.x;       // 0..127
    int cnt = g_lv.cnt[level];
    int tr = j / cnt, pos = j % cnt;
    int node = tr * NODES_PER_TREE + g_lv.idx[level][pos];
    const float* c = Cf + (size_t)j * HID2;
    const float* w = Wt + (size_t)h * HID2;
    float s = xvx[(size_t)j * HID + h] + bias[h];
#pragma unroll 4
    for (int l = 0; l < HID2; ++l) s += c[l] * w[l];
    H[(size_t)node * HID + h] = tanhf(s);
}

// out[n] = log_softmax(H[n] @ Wout + bout)
__global__ void k_out(const float* __restrict__ H, const float* __restrict__ Wo,
                      const float* __restrict__ bo, float* __restrict__ out) {
    int n = blockIdx.x * blockDim.x + threadIdx.x;
    if (n >= N_NODES) return;
    const float* hrow = H + (size_t)n * HID;
    float lg[OUT_DIM];
#pragma unroll
    for (int o = 0; o < OUT_DIM; ++o) lg[o] = bo[o];
    for (int k = 0; k < HID; ++k) {
        float hv = hrow[k];
#pragma unroll
        for (int o = 0; o < OUT_DIM; ++o) lg[o] += hv * Wo[k * OUT_DIM + o];
    }
    float m = lg[0];
#pragma unroll
    for (int o = 1; o < OUT_DIM; ++o) m = fmaxf(m, lg[o]);
    float se = 0.f;
#pragma unroll
    for (int o = 0; o < OUT_DIM; ++o) se += expf(lg[o] - m);
    float lse = m + logf(se);
#pragma unroll
    for (int o = 0; o < OUT_DIM; ++o) out[(size_t)n * OUT_DIM + o] = lg[o] - lse;
}

// ---------------------------------------------------------------------------
extern "C" void kernel_launch(void* const* d_in, const int* in_sizes, int n_in,
                              void* d_out, int out_size, void* d_ws, size_t ws_size,
                              hipStream_t stream) {
    const float* embed = (const float*)d_in[0];
    const float* V     = (const float*)d_in[1];
    const float* W     = (const float*)d_in[2];
    const float* bias  = (const float*)d_in[3];
    const float* Wo    = (const float*)d_in[4];
    const float* bo    = (const float*)d_in[5];
    const int*   word  = (const int*)d_in[7];   // word_idx (is_leaf/left/right are baked in)
    float* out = (float*)d_out;
    (void)in_sizes; (void)n_in; (void)out_size;

    // Carve workspace (~19 MB total)
    char* ws = (char*)d_ws;
    size_t off = 0;
    auto carve = [&](size_t bytes) -> char* {
        char* p = ws + off;
        off += (bytes + 255) & ~(size_t)255;
        return p;
    };
    __bf16* Vbf = (__bf16*)carve((size_t)HID * HID2 * HID2 * sizeof(__bf16)); // 16.8 MB
    float*  H   = (float*) carve((size_t)N_NODES * HID * sizeof(float));      // 1.04 MB
    float*  Wt  = (float*) carve((size_t)HID * HID2 * sizeof(float));
    float*  Cf  = (float*) carve((size_t)512 * HID2 * sizeof(float));
    __bf16* Cbf = (__bf16*)carve((size_t)512 * HID2 * sizeof(__bf16));
    float*  xvx = (float*) carve((size_t)512 * HID * sizeof(float));
    if (off > ws_size) return;  // workspace too small: refuse to scribble OOB

    const int nV = HID * HID2 * HID2;
    k_cvt_bf16<<<(nV + 255) / 256, 256, 0, stream>>>(V, Vbf, nV);
    k_wt<<<(HID2 * HID + 255) / 256, 256, 0, stream>>>(W, Wt);
    k_leaf<<<N_NODES, HID, 0, stream>>>(embed, word, H);

    for (int level = 1; level <= MAX_H; ++level) {
        int B = N_TREES * (N_LEAVES >> level);  // 512,256,128,64,32,16
        int numTiles = B / 16;                  // 32,16,8,4,2,1
        k_gather<<<B, HID2, 0, stream>>>(H, Cf, Cbf, level);
        dim3 g((numTiles + TILES_PER_BLOCK - 1) / TILES_PER_BLOCK, HID);
        k_bilinear<<<g, 128, 0, stream>>>(Vbf, Cbf, Cf, xvx, numTiles);
        k_finalize<<<B, HID, 0, stream>>>(xvx, Cf, Wt, bias, H, level);
    }
    k_out<<<(N_NODES + 255) / 256, 256, 0, stream>>>(H, Wo, bo, out);
}
